// IMPContext_76012331205213
// MI455X (gfx1250) — compile-verified, matrix operand order checked by reading, not run on hardware
//
#include <hip/hip_runtime.h>
#include <hip/hip_bf16.h>
#include <math.h>

// ---------------------------------------------------------------- constants
#define NOBJ 2048
#define NREL 32768
#define ODIM 4096
#define POOLD 2048
#define HDIM 512
#define EDIM 200
#define OBJIN_K 4424     // ODIM + EDIM + 128
#define AUGM_K  4808     // EDIM + ODIM + HDIM

typedef __attribute__((ext_vector_type(16))) __bf16 v16bf;
typedef __attribute__((ext_vector_type(8)))  float  v8f;

// f32 -> bf16 via native conversion hardware (pairs fuse to v_cvt_pk_bf16_f32).
__device__ __forceinline__ __bf16 f2bf(float f) {
    return (__bf16)f;
}

// Fast-path fragment load: 16 bf16 from two contiguous 8-float chunks.
// Elements 0..7  = K = kb .. kb+7
// Elements 8..15 = K = kb+16 .. kb+23   (kb = k0 + (lane>=16 ? 8 : 0))
__device__ __forceinline__ v16bf load16_full(const float* __restrict__ rowp, int kb) {
    const float4* p = (const float4*)(rowp + kb);
    const float4* q = (const float4*)(rowp + kb + 16);
    float4 a0 = p[0], a1 = p[1], b0 = q[0], b1 = q[1];
    v16bf r;
    r[0]=f2bf(a0.x); r[1]=f2bf(a0.y); r[2]=f2bf(a0.z); r[3]=f2bf(a0.w);
    r[4]=f2bf(a1.x); r[5]=f2bf(a1.y); r[6]=f2bf(a1.z); r[7]=f2bf(a1.w);
    r[8]=f2bf(b0.x); r[9]=f2bf(b0.y); r[10]=f2bf(b0.z); r[11]=f2bf(b0.w);
    r[12]=f2bf(b1.x); r[13]=f2bf(b1.y); r[14]=f2bf(b1.z); r[15]=f2bf(b1.w);
    return r;
}

// Tail fragment load with per-element K guard (used at most once per wave).
__device__ __forceinline__ v16bf load16_tail(const float* __restrict__ rowp,
                                             int kb, int K) {
    v16bf r;
    #pragma unroll
    for (int j = 0; j < 8; ++j) {
        int ka = kb + j, kc = kb + 16 + j;
        r[j]   = (ka < K) ? f2bf(rowp[ka]) : f2bf(0.f);
        r[j+8] = (kc < K) ? f2bf(rowp[kc]) : f2bf(0.f);
    }
    return r;
}

// ---------------------------------------------------------------- GEMM
// C[M,N] = act( A[M,K] @ W[N,K]^T + bias ) (+ res)
// One wave computes a 16x64 strip (4 WMMA tiles sharing the A fragment).
// Tiles are linearized N-fastest so a 16-row band of A plus the whole weight
// matrix stay resident in L2 while all N-tiles consume them.
#define GWAVES 4
__global__ void __launch_bounds__(128)
k_gemm(const float* __restrict__ A, const float* __restrict__ W,
       const float* __restrict__ bias, const float* __restrict__ res,
       float* __restrict__ C, int M, int N, int K, int do_relu, int tiles)
{
    int lane = threadIdx.x & 31;
    int wv   = threadIdx.x >> 5;
    int t    = blockIdx.x * GWAVES + wv;           // wave-uniform
    if (t >= tiles) return;                        // whole wave exits together
    int ntn  = N >> 6;                             // N / 64 macro-cols
    int tm   = t / ntn;
    int tn   = t - tm * ntn;
    int l15  = lane & 15;
    int rowA = (tm << 4) + l15;
    int wb   = tn << 6;
    int koff = (lane >> 4) << 3;                   // 0 or 8
    const float* arow = A + (size_t)rowA * K;
    const float* wr0  = W + (size_t)(wb + l15) * K;
    const float* wr1  = wr0 + (size_t)16 * K;
    const float* wr2  = wr0 + (size_t)32 * K;
    const float* wr3  = wr0 + (size_t)48 * K;

    v8f acc0 = {}, acc1 = {}, acc2 = {}, acc3 = {};
    int kfull = K & ~31;                           // branch-free main loop
    for (int k0 = 0; k0 < kfull; k0 += 32) {
        int kb = k0 + koff;
        __builtin_prefetch(arow + k0 + 256, 0, 1);
        __builtin_prefetch(wr0  + k0 + 256, 0, 1);
        v16bf a  = load16_full(arow, kb);
        v16bf b0 = load16_full(wr0,  kb);
        v16bf b1 = load16_full(wr1,  kb);
        v16bf b2 = load16_full(wr2,  kb);
        v16bf b3 = load16_full(wr3,  kb);
        acc0 = __builtin_amdgcn_wmma_f32_16x16x32_bf16(false, a, false, b0, (short)0, acc0, false, false);
        acc1 = __builtin_amdgcn_wmma_f32_16x16x32_bf16(false, a, false, b1, (short)0, acc1, false, false);
        acc2 = __builtin_amdgcn_wmma_f32_16x16x32_bf16(false, a, false, b2, (short)0, acc2, false, false);
        acc3 = __builtin_amdgcn_wmma_f32_16x16x32_bf16(false, a, false, b3, (short)0, acc3, false, false);
    }
    if (kfull < K) {                               // single guarded tail step
        int kb = kfull + koff;
        v16bf a  = load16_tail(arow, kb, K);
        v16bf b0 = load16_tail(wr0,  kb, K);
        v16bf b1 = load16_tail(wr1,  kb, K);
        v16bf b2 = load16_tail(wr2,  kb, K);
        v16bf b3 = load16_tail(wr3,  kb, K);
        acc0 = __builtin_amdgcn_wmma_f32_16x16x32_bf16(false, a, false, b0, (short)0, acc0, false, false);
        acc1 = __builtin_amdgcn_wmma_f32_16x16x32_bf16(false, a, false, b1, (short)0, acc1, false, false);
        acc2 = __builtin_amdgcn_wmma_f32_16x16x32_bf16(false, a, false, b2, (short)0, acc2, false, false);
        acc3 = __builtin_amdgcn_wmma_f32_16x16x32_bf16(false, a, false, b3, (short)0, acc3, false, false);
    }
    // D layout: VGPR i -> M = i (lanes 0-15) / i+8 (lanes 16-31); N = lane&15
    int rbase = (tm << 4) + ((lane >> 4) << 3);
    #pragma unroll
    for (int i = 0; i < 8; ++i) {
        int r = rbase + i;
        size_t ro = (size_t)r * N;
        #pragma unroll
        for (int j = 0; j < 4; ++j) {
            int col = wb + (j << 4) + l15;
            float v;
            switch (j) { case 0: v = acc0[i]; break; case 1: v = acc1[i]; break;
                         case 2: v = acc2[i]; break; default: v = acc3[i]; break; }
            v += bias[col];
            if (do_relu) v = fmaxf(v, 0.f);
            if (res) v += res[ro + col];
            C[ro + col] = v;
        }
    }
}

// ---------------------------------------------------------------- small kernels
__global__ void k_pos(const float* __restrict__ boxes,
                      const float* __restrict__ p1W, const float* __restrict__ p1b,
                      const float* __restrict__ bg, const float* __restrict__ bb,
                      const float* __restrict__ bm, const float* __restrict__ bv,
                      const float* __restrict__ p2W, const float* __restrict__ p2b,
                      float* __restrict__ pos_embed)
{
    int o = blockIdx.x;
    __shared__ float sp[32];
    float x1 = boxes[o*4+0], y1 = boxes[o*4+1], x2 = boxes[o*4+2], y2 = boxes[o*4+3];
    float w = x2 - x1 + 1.f, h = y2 - y1 + 1.f;
    float cx = x1 + 0.5f*w, cy = y1 + 0.5f*h;
    float f[9] = { w/1024.f, h/768.f, cx/1024.f, cy/768.f,
                   x1/1024.f, y1/768.f, x2/1024.f, y2/768.f, w*h/(1024.f*768.f) };
    int t = threadIdx.x;
    if (t < 32) {
        float s = p1b[t];
        #pragma unroll
        for (int j = 0; j < 9; ++j) s += f[j] * p1W[t*9 + j];
        sp[t] = (s - bm[t]) * bg[t] * rsqrtf(bv[t] + 1e-5f) + bb[t];
    }
    __syncthreads();
    float s = p2b[t];
    #pragma unroll 8
    for (int j = 0; j < 32; ++j) s += sp[j] * p2W[t*32 + j];
    pos_embed[(size_t)o*128 + t] = fmaxf(s, 0.f);
}

__global__ void k_objin(const float* __restrict__ inst, const float* __restrict__ embd,
                        const int* __restrict__ lab, const float* __restrict__ pos,
                        float* __restrict__ out)
{
    size_t idx = (size_t)blockIdx.x * blockDim.x + threadIdx.x;
    if (idx >= (size_t)NOBJ * OBJIN_K) return;
    int o = (int)(idx / OBJIN_K), c = (int)(idx % OBJIN_K);
    float v;
    if (c < ODIM)            v = inst[(size_t)o*ODIM + c];
    else if (c < ODIM+EDIM)  v = embd[(size_t)lab[o]*EDIM + (c - ODIM)];
    else                     v = pos[(size_t)o*128 + (c - ODIM - EDIM)];
    out[idx] = v;
}

__global__ void k_augment(const float* __restrict__ inst, const float* __restrict__ embl,
                          const int* __restrict__ lab, const float* __restrict__ aug,
                          float* __restrict__ out)
{
    size_t idx = (size_t)blockIdx.x * blockDim.x + threadIdx.x;
    if (idx >= (size_t)NOBJ * AUGM_K) return;
    int o = (int)(idx / AUGM_K), c = (int)(idx % AUGM_K);
    float v;
    if (c < EDIM)            v = embl[(size_t)lab[o]*EDIM + c];
    else if (c < EDIM+ODIM)  v = inst[(size_t)o*ODIM + (c - EDIM)];
    else                     v = aug[(size_t)o*HDIM + (c - EDIM - ODIM)];
    out[idx] = v;
}

__global__ void k_pairgather(const float* __restrict__ fused, const int* __restrict__ sub,
                             const int* __restrict__ obj, float* __restrict__ pair)
{
    size_t idx = (size_t)blockIdx.x * blockDim.x + threadIdx.x;
    if (idx >= (size_t)NREL * 1024) return;
    int r = (int)(idx >> 10), c = (int)(idx & 1023);
    int src = (c < 512) ? sub[r] : obj[r];
    pair[idx] = fused[(size_t)src * 1024 + c];
}

__device__ __forceinline__ void boxinfo8(float x1, float y1, float x2, float y2, float* o) {
    float w = x2 - x1 + 1.f, h = y2 - y1 + 1.f;
    o[0]=x1; o[1]=y1; o[2]=x2; o[3]=y2;
    o[4]=x1 + 0.5f*w; o[5]=y1 + 0.5f*h; o[6]=w; o[7]=h;
}

__global__ void k_geo(const float* __restrict__ boxes, const int* __restrict__ sub,
                      const int* __restrict__ obj, float* __restrict__ geo)
{
    int r = blockIdx.x * blockDim.x + threadIdx.x;
    if (r >= NREL) return;
    const float inv = 1.f / 1024.f;
    int s = sub[r], o = obj[r];
    float b1[8], b2[8], g[32];
    boxinfo8(boxes[s*4], boxes[s*4+1], boxes[s*4+2], boxes[s*4+3], b1);
    boxinfo8(boxes[o*4], boxes[o*4+1], boxes[o*4+2], boxes[o*4+3], b2);
    #pragma unroll
    for (int i = 0; i < 8; ++i) { b1[i] *= inv; b2[i] *= inv; g[i] = b1[i]; g[8+i] = b2[i]; }
    float ux1 = fminf(b1[0], b2[0]), uy1 = fminf(b1[1], b2[1]);
    float ux2 = fmaxf(b1[2], b2[2]), uy2 = fmaxf(b1[3], b2[3]);
    float ix1 = fmaxf(b1[0], b2[0]), iy1 = fmaxf(b1[1], b2[1]);
    float ix2 = fminf(b1[2], b2[2]), iy2 = fminf(b1[3], b2[3]);
    float ub[8], ib[8];
    boxinfo8(ux1, uy1, ux2, uy2, ub);
    boxinfo8(ix1, iy1, ix2, iy2, ib);
    bool bad = (ix2 < ix1) || (iy2 < iy1);
    #pragma unroll
    for (int i = 0; i < 8; ++i) { g[16+i] = ub[i]; g[24+i] = bad ? 0.f : ib[i]; }
    #pragma unroll
    for (int i = 0; i < 32; ++i) geo[(size_t)r*32 + i] = g[i];
}

__global__ void k_mul_inplace(float* __restrict__ a, const float* __restrict__ b, size_t n) {
    size_t idx = (size_t)blockIdx.x * blockDim.x + threadIdx.x;
    if (idx < n) a[idx] *= b[idx];
}

__global__ void k_zero(float* __restrict__ p, size_t n) {
    size_t idx = (size_t)blockIdx.x * blockDim.x + threadIdx.x;
    if (idx < n) p[idx] = 0.f;
}

__global__ void k_copy(float* __restrict__ d, const float* __restrict__ s, size_t n) {
    size_t idx = (size_t)blockIdx.x * blockDim.x + threadIdx.x;
    if (idx < n) d[idx] = s[idx];
}

// per-relation gate weights ws/wo (wave-per-relation dot over 1024)
__global__ void __launch_bounds__(128)
k_wswo(const float* __restrict__ vert, const float* __restrict__ edge,
       const int* __restrict__ sub, const int* __restrict__ obj,
       const float* __restrict__ swW, const float* __restrict__ swb,
       const float* __restrict__ owW, const float* __restrict__ owb,
       float* __restrict__ wsv, float* __restrict__ wov)
{
    int lane = threadIdx.x & 31, wv = threadIdx.x >> 5;
    int r = blockIdx.x * 4 + wv;
    if (r >= NREL) return;
    const float* sv = vert + (size_t)sub[r] * HDIM;
    const float* ov = vert + (size_t)obj[r] * HDIM;
    const float* ed = edge + (size_t)r * HDIM;
    float s1 = 0.f, s2 = 0.f;
    for (int c = lane; c < HDIM; c += 32) {
        float e = ed[c];
        s1 += sv[c] * swW[c] + e * swW[HDIM + c];
        s2 += ov[c] * owW[c] + e * owW[HDIM + c];
    }
    #pragma unroll
    for (int off = 16; off > 0; off >>= 1) {
        s1 += __shfl_down(s1, off, 32);
        s2 += __shfl_down(s2, off, 32);
    }
    if (lane == 0) {
        wsv[r] = 1.f / (1.f + __expf(-(s1 + swb[0])));
        wov[r] = 1.f / (1.f + __expf(-(s2 + owb[0])));
    }
}

__global__ void k_msg(const float* __restrict__ vert, const int* __restrict__ sub,
                      const int* __restrict__ obj, const float* __restrict__ wsv,
                      const float* __restrict__ wov, float* __restrict__ msg)
{
    size_t idx = (size_t)blockIdx.x * blockDim.x + threadIdx.x;
    if (idx >= (size_t)NREL * HDIM) return;
    int r = (int)(idx >> 9), c = (int)(idx & 511);
    msg[idx] = wsv[r] * vert[(size_t)sub[r]*HDIM + c] + wov[r] * vert[(size_t)obj[r]*HDIM + c];
}

__global__ void k_scatter(const float* __restrict__ edge, const float* __restrict__ wsv,
                          const float* __restrict__ wov, const int* __restrict__ sub,
                          const int* __restrict__ obj, float* __restrict__ ctx)
{
    size_t idx = (size_t)blockIdx.x * blockDim.x + threadIdx.x;
    if (idx >= (size_t)NREL * HDIM) return;
    int r = (int)(idx >> 9), c = (int)(idx & 511);
    float e = edge[idx];
    atomicAdd(&ctx[(size_t)sub[r]*HDIM + c], wsv[r] * e);
    atomicAdd(&ctx[(size_t)obj[r]*HDIM + c], wov[r] * e);
}

// GRU gate combine; gh==nullptr -> gates use bhh broadcast and h==0
__global__ void k_gru_gate(const float* __restrict__ gi, const float* __restrict__ gh,
                           const float* __restrict__ bhh, const float* __restrict__ h,
                           float* __restrict__ out, int M)
{
    size_t idx = (size_t)blockIdx.x * blockDim.x + threadIdx.x;
    if (idx >= (size_t)M * HDIM) return;
    int m = (int)(idx >> 9), c = (int)(idx & 511);
    size_t g = (size_t)m * (3 * HDIM);
    float ir = gi[g + c], iz = gi[g + HDIM + c], in_ = gi[g + 2*HDIM + c];
    float hr, hz, hn;
    if (gh) { hr = gh[g + c]; hz = gh[g + HDIM + c]; hn = gh[g + 2*HDIM + c]; }
    else    { hr = bhh[c];    hz = bhh[HDIM + c];    hn = bhh[2*HDIM + c]; }
    float hv = h ? h[(size_t)m*HDIM + c] : 0.f;
    float rr = 1.f / (1.f + __expf(-(ir + hr)));
    float zz = 1.f / (1.f + __expf(-(iz + hz)));
    float nn = tanhf(in_ + rr * hn);
    out[(size_t)m*HDIM + c] = (1.f - zz) * nn + zz * hv;
}

// ---------------------------------------------------------------- host
extern "C" void kernel_launch(void* const* d_in, const int* in_sizes, int n_in,
                              void* d_out, int out_size, void* d_ws, size_t ws_size,
                              hipStream_t stream)
{
    (void)in_sizes; (void)n_in; (void)out_size; (void)ws_size;
    const float* inst  = (const float*)d_in[0];
    const float* uni   = (const float*)d_in[1];
    const float* boxes = (const float*)d_in[2];
    const float* embd  = (const float*)d_in[3];
    const float* embl  = (const float*)d_in[4];
    const float* upW   = (const float*)d_in[5];  const float* upb  = (const float*)d_in[6];
    const float* puW   = (const float*)d_in[7];  const float* pub  = (const float*)d_in[8];
    const float* p1W   = (const float*)d_in[9];  const float* p1b  = (const float*)d_in[10];
    const float* bg    = (const float*)d_in[11]; const float* bb   = (const float*)d_in[12];
    const float* bm    = (const float*)d_in[13]; const float* bv   = (const float*)d_in[14];
    const float* p2W   = (const float*)d_in[15]; const float* p2b  = (const float*)d_in[16];
    const float* s1W   = (const float*)d_in[17]; const float* s1b  = (const float*)d_in[18];
    const float* s2W   = (const float*)d_in[19]; const float* s2b  = (const float*)d_in[20];
    const float* pfW   = (const float*)d_in[21]; const float* pfb  = (const float*)d_in[22];
    const float* ohW   = (const float*)d_in[23]; const float* ohb  = (const float*)d_in[24];
    const float* ofW   = (const float*)d_in[25]; const float* ofb  = (const float*)d_in[26];
    const float* oUW   = (const float*)d_in[27]; const float* oUb  = (const float*)d_in[28];
    const float* eUW   = (const float*)d_in[29]; const float* eUb  = (const float*)d_in[30];
    const float* eWih  = (const float*)d_in[31]; const float* eWhh = (const float*)d_in[32];
    const float* ebih  = (const float*)d_in[33]; const float* ebhh = (const float*)d_in[34];
    const float* nWih  = (const float*)d_in[35]; const float* nWhh = (const float*)d_in[36];
    const float* nbih  = (const float*)d_in[37]; const float* nbhh = (const float*)d_in[38];
    const float* swW   = (const float*)d_in[39]; const float* swb  = (const float*)d_in[40];
    const float* owW   = (const float*)d_in[41]; const float* owb  = (const float*)d_in[42];
    const int*   lab   = (const int*)d_in[43];
    const int*   sub   = (const int*)d_in[44];
    const int*   obj   = (const int*)d_in[45];

    float* wsp = (float*)d_ws;
    size_t off = 0;
    auto take = [&](size_t n) { float* p = wsp + off; off += n; return p; };
    float* pos_embed = take((size_t)NOBJ * 128);
    float* objin     = take((size_t)NOBJ * OBJIN_K);
    float* aug       = take((size_t)NOBJ * HDIM);
    float* augment   = take((size_t)NOBJ * AUGM_K);
    float* fused     = take((size_t)NOBJ * 1024);
    float* pair      = take((size_t)NREL * 1024);
    float* geo       = take((size_t)NREL * 32);
    float* spt_h     = take((size_t)NREL * HDIM);
    float* spt       = take((size_t)NREL * 1024);
    float* rel       = take((size_t)NREL * POOLD);
    float* objf      = take((size_t)NOBJ * POOLD);
    float* objrep    = take((size_t)NOBJ * HDIM);
    float* relrep    = take((size_t)NREL * HDIM);
    float* vertA     = take((size_t)NOBJ * HDIM);
    float* vertB     = take((size_t)NOBJ * HDIM);
    float* edgeA     = take((size_t)NREL * HDIM);
    float* edgeB     = take((size_t)NREL * HDIM);
    float* gi_e      = take((size_t)NREL * 3 * HDIM);
    float* gh_e      = take((size_t)NREL * 3 * HDIM);
    float* gi_n      = take((size_t)NOBJ * 3 * HDIM);
    float* gh_n      = take((size_t)NOBJ * 3 * HDIM);
    float* wsv       = take((size_t)NREL);
    float* wov       = take((size_t)NREL);
    float* msg       = take((size_t)NREL * HDIM);
    float* ctx       = take((size_t)NOBJ * HDIM);

    auto gemm = [&](const float* A, const float* Wt, const float* b, const float* res,
                    float* C, int M, int N, int K, int relu) {
        int tiles  = (M / 16) * (N / 64);
        int blocks = (tiles + GWAVES - 1) / GWAVES;
        k_gemm<<<blocks, 128, 0, stream>>>(A, Wt, b, res, C, M, N, K, relu, tiles);
    };
    auto ew = [&](size_t total) { return (unsigned)((total + 255) / 256); };

    // 1. positional embedding
    k_pos<<<NOBJ, 128, 0, stream>>>(boxes, p1W, p1b, bg, bb, bm, bv, p2W, p2b, pos_embed);
    // 2. objin = [inst | emb_dist | pos_embed]
    k_objin<<<ew((size_t)NOBJ*OBJIN_K), 256, 0, stream>>>(inst, embd, lab, pos_embed, objin);
    // 3. aug = objin @ objhid^T + b
    gemm(objin, ohW, ohb, nullptr, aug, NOBJ, HDIM, OBJIN_K, 0);
    // 4. augment = [emb_label | inst | aug]
    k_augment<<<ew((size_t)NOBJ*AUGM_K), 256, 0, stream>>>(inst, embl, lab, aug, augment);
    // 5. fused = augment @ pairup^T + b
    gemm(augment, puW, pub, nullptr, fused, NOBJ, 1024, AUGM_K, 0);
    // 6. pair gather
    k_pairgather<<<ew((size_t)NREL*1024), 256, 0, stream>>>(fused, sub, obj, pair);
    // 7. geometry features
    k_geo<<<ew(NREL), 256, 0, stream>>>(boxes, sub, obj, geo);
    // 8-9. spt = relu(relu(geo@spt1)@spt2)
    gemm(geo, s1W, s1b, nullptr, spt_h, NREL, HDIM, 32, 1);
    gemm(spt_h, s2W, s2b, nullptr, spt, NREL, 1024, HDIM, 1);
    // 10. pair *= spt
    k_mul_inplace<<<ew((size_t)NREL*1024), 256, 0, stream>>>(pair, spt, (size_t)NREL*1024);
    // 11. rel = union @ up^T + b   (the 550 GFLOP GEMM)
    gemm(uni, upW, upb, nullptr, rel, NREL, POOLD, ODIM, 0);
    // 12. rel = relu(pair @ pairfin^T + b) + rel
    gemm(pair, pfW, pfb, rel, rel, NREL, POOLD, 1024, 1);
    // 13. obj_feats = relu(augment @ objfin^T + b)
    gemm(augment, ofW, ofb, nullptr, objf, NOBJ, POOLD, AUGM_K, 1);
    // 14. obj_rep = obj_feats @ objU^T + b
    gemm(objf, oUW, oUb, nullptr, objrep, NOBJ, HDIM, POOLD, 0);
    // 15. rel_rep = relu(rel @ edgeU^T + b)
    gemm(rel, eUW, eUb, nullptr, relrep, NREL, HDIM, POOLD, 1);
    // 16. vert = GRU(obj_rep, 0)
    gemm(objrep, nWih, nbih, nullptr, gi_n, NOBJ, 3*HDIM, HDIM, 0);
    k_gru_gate<<<ew((size_t)NOBJ*HDIM), 256, 0, stream>>>(gi_n, nullptr, nbhh, nullptr, vertA, NOBJ);
    // 17. edge = GRU(rel_rep, 0)
    gemm(relrep, eWih, ebih, nullptr, gi_e, NREL, 3*HDIM, HDIM, 0);
    k_gru_gate<<<ew((size_t)NREL*HDIM), 256, 0, stream>>>(gi_e, nullptr, ebhh, nullptr, edgeA, NREL);

    // 18. message passing, 3 iterations (host-unrolled, pointer swap)
    float* vert = vertA; float* vert_n = vertB;
    float* edge = edgeA; float* edge_n = edgeB;
    for (int it = 0; it < 3; ++it) {
        k_wswo<<<(NREL + 3) / 4, 128, 0, stream>>>(vert, edge, sub, obj, swW, swb, owW, owb, wsv, wov);
        k_msg<<<ew((size_t)NREL*HDIM), 256, 0, stream>>>(vert, sub, obj, wsv, wov, msg);
        // new_edge = GRU(msg, edge)
        gemm(msg,  eWih, ebih, nullptr, gi_e, NREL, 3*HDIM, HDIM, 0);
        gemm(edge, eWhh, ebhh, nullptr, gh_e, NREL, 3*HDIM, HDIM, 0);
        k_gru_gate<<<ew((size_t)NREL*HDIM), 256, 0, stream>>>(gi_e, gh_e, ebhh, edge, edge_n, NREL);
        // ctx = segsum(ws*edge, sub) + segsum(wo*edge, obj)   (old edge)
        k_zero<<<ew((size_t)NOBJ*HDIM), 256, 0, stream>>>(ctx, (size_t)NOBJ*HDIM);
        k_scatter<<<ew((size_t)NREL*HDIM), 256, 0, stream>>>(edge, wsv, wov, sub, obj, ctx);
        // vert = GRU(ctx, vert)
        gemm(ctx,  nWih, nbih, nullptr, gi_n, NOBJ, 3*HDIM, HDIM, 0);
        gemm(vert, nWhh, nbhh, nullptr, gh_n, NOBJ, 3*HDIM, HDIM, 0);
        k_gru_gate<<<ew((size_t)NOBJ*HDIM), 256, 0, stream>>>(gi_n, gh_n, nbhh, vert, vert_n, NOBJ);
        float* tv = vert; vert = vert_n; vert_n = tv;
        float* te = edge; edge = edge_n; edge_n = te;
    }

    // 19. outputs: (vert, edge) concatenated
    float* out = (float*)d_out;
    k_copy<<<ew((size_t)NOBJ*HDIM), 256, 0, stream>>>(out, vert, (size_t)NOBJ*HDIM);
    k_copy<<<ew((size_t)NREL*HDIM), 256, 0, stream>>>(out + (size_t)NOBJ*HDIM, edge, (size_t)NREL*HDIM);
}
